// STPTLight_9285719294256
// MI455X (gfx1250) — matrix operand order repaired and checked by reading, not compiled, not run on hardware
//
#include <hip/hip_runtime.h>
#include <hip/hip_bf16.h>
#include <math.h>

typedef __attribute__((ext_vector_type(16))) __bf16 bf16x16;
typedef __attribute__((ext_vector_type(8)))  __bf16 bf16x8;
typedef __attribute__((ext_vector_type(8)))  float  f32x8;
typedef __attribute__((ext_vector_type(2)))  float  f32x2;
typedef __attribute__((ext_vector_type(4)))  int    i32x4;

// Problem constants
static constexpr int Bsz = 4;
static constexpr int Tsz = 1024;
static constexpr int Dsz = 1024;
static constexpr int Hsz = 16;
static constexpr int DKs = 64;

// ---------------------------------------------------------------------------
// Async global -> LDS copy of 16 bytes (GLOBAL_LOAD_ASYNC_TO_LDS_B128,
// tracked with ASYNCcnt). Builtin signature wants <4 x i32> pointers:
// arg0 = global (AS1), arg1 = LDS (AS3), then immediate offset and cpol.
// ---------------------------------------------------------------------------
typedef __attribute__((address_space(1))) i32x4 as1_i32x4;
typedef __attribute__((address_space(3))) i32x4 as3_i32x4;

__device__ __forceinline__ void async_copy16(const __bf16* g, __bf16* l) {
    __builtin_amdgcn_global_load_async_to_lds_b128(
        (as1_i32x4*)(g), (as3_i32x4*)(l), 0, 0);
}
__device__ __forceinline__ void wait_async0() {
    asm volatile("s_wait_asynccnt 0x0" ::: "memory");
}

// ---------------------------------------------------------------------------
// f32 -> bf16 conversion
// ---------------------------------------------------------------------------
__global__ void cvt_f32_bf16(const float* __restrict__ s, __bf16* __restrict__ d, int n) {
    int i = blockIdx.x * blockDim.x + threadIdx.x;
    if (i < n) d[i] = (__bf16)s[i];
}

// ---------------------------------------------------------------------------
// C(M x N) = A(M x K, bf16 row-major) * B^T where Bt is stored [n][k]
// (i.e. the original torch Linear weight layout W[n][k]).
// Block tile 128x128, BK=32 (= one wmma_f32_16x16x32_bf16 K step).
// 8 waves: 2 (M) x 4 (N); each wave: 4x2 grid of 16x16 f32 accumulators.
// Double-buffered LDS staging via async global->LDS copies.
// ---------------------------------------------------------------------------
template <bool OUTF32>
__global__ __launch_bounds__(256)
void gemm_bt_bf16(const __bf16* __restrict__ A, const __bf16* __restrict__ Bt,
                  float* __restrict__ Cf, __bf16* __restrict__ Cb,
                  int M, int N, int K) {
    constexpr int BM = 128, BN = 128, BK = 32;
    __shared__ __bf16 As[2][BM * BK];
    __shared__ __bf16 Bs[2][BN * BK];

    const int tid  = threadIdx.x;
    const int lane = tid & 31;
    const int wave = tid >> 5;
    const int wr   = wave >> 2;   // 0..1  (64-row band)
    const int wc   = wave & 3;    // 0..3  (32-col band)
    const int sel  = lane >> 4;   // lane half
    const int l16  = lane & 15;
    const int m0   = blockIdx.y * BM;
    const int n0   = blockIdx.x * BN;

    f32x8 acc[4][2];
#pragma unroll
    for (int mi = 0; mi < 4; ++mi)
#pragma unroll
        for (int ni = 0; ni < 2; ++ni) acc[mi][ni] = {};

    // Issue one tile's worth of async copies: 128 rows x 32 bf16 each of A and
    // Bt = 512 16-byte chunks per matrix; 2 chunks per thread per matrix.
    auto issue_loads = [&](int buf, int k0) {
#pragma unroll
        for (int i = 0; i < 2; ++i) {
            int ch  = tid * 2 + i;           // 0..511
            int row = ch >> 2;
            int ko  = (ch & 3) * 8;
            async_copy16(&A [(size_t)(m0 + row) * K + k0 + ko], &As[buf][row * BK + ko]);
            async_copy16(&Bt[(size_t)(n0 + row) * K + k0 + ko], &Bs[buf][row * BK + ko]);
        }
    };

    const int nsteps = K / BK;
    issue_loads(0, 0);

    for (int s = 0; s < nsteps; ++s) {
        const int cur = s & 1;
        wait_async0();          // this wave's pending tile copies are complete
        __syncthreads();        // every wave's copies complete; prev reads done
        if (s + 1 < nsteps) issue_loads(cur ^ 1, (s + 1) * BK);

        const __bf16* Ab = As[cur];
        const __bf16* Bb = Bs[cur];

        // Gather WMMA fragments from LDS (contiguous 16B reads per lane).
        bf16x16 afr[4];
        bf16x16 bfr[2];
#pragma unroll
        for (int mi = 0; mi < 4; ++mi) {
            const __bf16* p = &Ab[(wr * 64 + mi * 16 + l16) * BK];
#pragma unroll
            for (int j = 0; j < 8; ++j) {
                afr[mi][j]     = p[sel * 8 + j];        // K = sel*8 + j
                afr[mi][8 + j] = p[16 + sel * 8 + j];   // K = 16 + sel*8 + j
            }
        }
#pragma unroll
        for (int ni = 0; ni < 2; ++ni) {
            const __bf16* p = &Bb[(wc * 32 + ni * 16 + l16) * BK];
#pragma unroll
            for (int j = 0; j < 16; ++j) bfr[ni][j] = p[(j < 8) ? (sel * 8 + j) : (16 + sel * 8 + (j - 8))]; // K = table order, same as A
        }

#pragma unroll
        for (int mi = 0; mi < 4; ++mi)
#pragma unroll
            for (int ni = 0; ni < 2; ++ni)
                acc[mi][ni] = __builtin_amdgcn_wmma_f32_16x16x32_bf16(
                    false, afr[mi], false, bfr[ni], (short)0, acc[mi][ni],
                    false, false);
        // fragments consumed by WMMA => DScnt drained before next barrier
    }

    // Epilogue: C fragment layout: elem j -> row (sel*8 + j), col l16.
#pragma unroll
    for (int mi = 0; mi < 4; ++mi)
#pragma unroll
        for (int ni = 0; ni < 2; ++ni) {
            int n = n0 + wc * 32 + ni * 16 + l16;
#pragma unroll
            for (int j = 0; j < 8; ++j) {
                int m = m0 + wr * 64 + mi * 16 + sel * 8 + j;
                if (OUTF32) Cf[(size_t)m * N + n] = acc[mi][ni][j];
                else        Cb[(size_t)m * N + n] = (__bf16)acc[mi][ni][j];
            }
        }
}

// ---------------------------------------------------------------------------
// Chunked gated recurrence. One wave per (b,h) chain; chunk C = 16.
//   y_t = a_t (q_t S0) + sum_{s<=t} c[t][s] (q_t . v_s) k_s
//   S'  = a_C S0 + sum_s d_s v_s k_s^T
// State S (64x64 f32) lives in 16 v8f register fragments; matmuls run on
// v_wmma_f32_16x16x4_f32 to keep the long-range state in f32.
// ---------------------------------------------------------------------------
__global__ __launch_bounds__(32)
void recurrence_kernel(const __bf16* __restrict__ qb, const __bf16* __restrict__ kb,
                       const __bf16* __restrict__ vb,
                       const float* __restrict__ Wg, const float* __restrict__ bg,
                       const float* __restrict__ Wl, const float* __restrict__ bl,
                       __bf16* __restrict__ yb) {
    constexpr int C = 16, NC = Tsz / C;
    const int bh  = blockIdx.x;
    const int bbi = bh / Hsz;
    const int hh  = bh % Hsz;
    const int lane = threadIdx.x;
    const int sel  = lane >> 4;
    const int l16  = lane & 15;

    __shared__ float Qs[C][DKs];
    __shared__ float Ks[C][DKs];
    __shared__ float Vs[C][DKs];
    __shared__ float Ss[DKs][DKs];   // state snapshot for B-fragment gathers
    __shared__ float Ps[C][C];
    __shared__ float rho[C], aa[C], dd[C], cm[C][C];

    const float wg = Wg[hh], bgv = bg[hh], wl = Wl[hh], blv = bl[hh];

    f32x8 S[4][4];
#pragma unroll
    for (int vi = 0; vi < 4; ++vi)
#pragma unroll
        for (int ki = 0; ki < 4; ++ki) S[vi][ki] = {};

    const size_t base = (size_t)bbi * Tsz * Dsz + (size_t)hh * DKs;

    for (int c = 0; c < NC; ++c) {
        const int t0 = c * C;
        // ---- load chunk (16x64 each of q,k,v) into LDS as f32 ----
        // lane -> (row = lane>>1, half = lane&1): 32 contiguous bf16 = 4x16B.
        {
            int row = lane >> 1, half = lane & 1;
            size_t g = base + (size_t)(t0 + row) * Dsz + half * 32;
            const bf16x8* gq = (const bf16x8*)&qb[g];
            const bf16x8* gk = (const bf16x8*)&kb[g];
            const bf16x8* gv = (const bf16x8*)&vb[g];
            // prefetch next chunk while this one is being processed
            __builtin_prefetch(&qb[g + (size_t)C * Dsz], 0, 3);
            __builtin_prefetch(&kb[g + (size_t)C * Dsz], 0, 3);
            __builtin_prefetch(&vb[g + (size_t)C * Dsz], 0, 3);
#pragma unroll
            for (int i = 0; i < 4; ++i) {
                bf16x8 tq = gq[i], tk = gk[i], tv = gv[i];
#pragma unroll
                for (int j = 0; j < 8; ++j) {
                    Qs[row][half * 32 + i * 8 + j] = (float)tq[j];
                    Ks[row][half * 32 + i * 8 + j] = (float)tk[j];
                    Vs[row][half * 32 + i * 8 + j] = (float)tv[j];
                }
            }
        }
        __syncthreads();
        // ---- normalize K rows, compute rho (lanes 0..15, one row each) ----
        if (lane < C) {
            int s = lane;
            float ss = 0.f;
            for (int d = 0; d < DKs; ++d) { float kv = Ks[s][d]; ss += kv * kv; }
            float inv = 1.f / fmaxf(sqrtf(ss), 1e-12f);
            float sum = 0.f;
            for (int d = 0; d < DKs; ++d) { float kv = Ks[s][d] * inv; Ks[s][d] = kv; sum += kv; }
            float kbar = sum * (1.f / DKs);
            float gma  = 1.f / (1.f + expf(-(wg * kbar + bgv)));
            float lam  = 1.f / (1.f + expf(-(wl * kbar + blv)));
            rho[s] = (1.f - lam) * gma;
        }
        __syncthreads();
        // ---- decay coefficients (lane t computes row t) ----
        if (lane < C) {
            int t = lane;
            float p = 1.f;
            for (int u = 0; u <= t; ++u) p *= rho[u];
            aa[t] = p;                              // a_t = prod_{u<=t} rho_u
            float pb = 1.f;
            for (int s = t; s >= 0; --s) {          // c[t][s] = (1-rho_s) prod_{s<u<=t} rho_u
                cm[t][s] = (1.f - rho[s]) * pb;
                pb *= rho[s];
            }
            for (int s = t + 1; s < C; ++s) cm[t][s] = 0.f;
            float pd = 1.f;
            for (int u = t + 1; u < C; ++u) pd *= rho[u];
            dd[t] = (1.f - rho[t]) * pd;            // d_s = (1-rho_s) prod_{u>s} rho_u
        }
        // ---- spill S registers -> LDS snapshot ----
#pragma unroll
        for (int vi = 0; vi < 4; ++vi)
#pragma unroll
            for (int ki = 0; ki < 4; ++ki)
#pragma unroll
                for (int j = 0; j < 8; ++j)
                    Ss[vi * 16 + sel * 8 + j][ki * 16 + l16] = S[vi][ki][j];
        __syncthreads();

        // ---- P = Q V^T (16x16, K=64) ----
        f32x8 p = {};
#pragma unroll
        for (int kk = 0; kk < 16; ++kk) {
            f32x2 a = { Qs[l16][kk * 4 + sel * 2 + 0], Qs[l16][kk * 4 + sel * 2 + 1] };
            f32x2 b = { Vs[l16][kk * 4 + sel * 2 + 0], Vs[l16][kk * 4 + sel * 2 + 1] };
            p = __builtin_amdgcn_wmma_f32_16x16x4_f32(false, a, false, b, (short)0, p, false, false);
        }
        // mask-scale P by c[t][s] and spill for reuse as an A-matrix
#pragma unroll
        for (int j = 0; j < 8; ++j) {
            int t = sel * 8 + j;
            Ps[t][l16] = p[j] * cm[t][l16];
        }
        __syncthreads();
        // fold a_t into Q rows (raw Q no longer needed)
        {
            int row = lane >> 1, half = lane & 1;
            float at = aa[row];
#pragma unroll
            for (int d = 0; d < 32; ++d) Qs[row][half * 32 + d] *= at;
        }
        __syncthreads();

        // ---- Y = (P.*C) K + (diag(a) Q) S0, 4 column tiles of 16 ----
#pragma unroll
        for (int ki = 0; ki < 4; ++ki) {
            f32x8 y = {};
#pragma unroll
            for (int kk = 0; kk < 4; ++kk) {      // intra: K = 16 (chunk pos s)
                int s0i = kk * 4 + sel * 2;
                f32x2 a = { Ps[l16][s0i + 0], Ps[l16][s0i + 1] };
                f32x2 b = { Ks[s0i + 0][ki * 16 + l16], Ks[s0i + 1][ki * 16 + l16] };
                y = __builtin_amdgcn_wmma_f32_16x16x4_f32(false, a, false, b, (short)0, y, false, false);
            }
#pragma unroll
            for (int kk = 0; kk < 16; ++kk) {     // inter: K = 64 (v dim of S0)
                int v0 = kk * 4 + sel * 2;
                f32x2 a = { Qs[l16][v0 + 0], Qs[l16][v0 + 1] };
                f32x2 b = { Ss[v0 + 0][ki * 16 + l16], Ss[v0 + 1][ki * 16 + l16] };
                y = __builtin_amdgcn_wmma_f32_16x16x4_f32(false, a, false, b, (short)0, y, false, false);
            }
#pragma unroll
            for (int j = 0; j < 8; ++j) {
                int t = sel * 8 + j;
                yb[base + (size_t)(t0 + t) * Dsz + ki * 16 + l16] = (__bf16)y[j];
            }
        }

        // ---- state update: S = a_C*S + (diag(d) V)^T K ----
        const float aC = aa[C - 1];
#pragma unroll
        for (int vi = 0; vi < 4; ++vi)
#pragma unroll
            for (int ki = 0; ki < 4; ++ki) {
                f32x8 cacc = S[vi][ki];
#pragma unroll
                for (int j = 0; j < 8; ++j) cacc[j] *= aC;
#pragma unroll
                for (int kk = 0; kk < 4; ++kk) {  // K = 16 (chunk pos s)
                    int s0i = kk * 4 + sel * 2;
                    f32x2 a = { dd[s0i + 0] * Vs[s0i + 0][vi * 16 + l16],
                                dd[s0i + 1] * Vs[s0i + 1][vi * 16 + l16] };
                    f32x2 b = { Ks[s0i + 0][ki * 16 + l16], Ks[s0i + 1][ki * 16 + l16] };
                    cacc = __builtin_amdgcn_wmma_f32_16x16x4_f32(false, a, false, b, (short)0, cacc, false, false);
                }
                S[vi][ki] = cacc;
            }
        __syncthreads();   // before next chunk overwrites LDS tiles
    }
}

// ---------------------------------------------------------------------------
// launch
// ---------------------------------------------------------------------------
extern "C" void kernel_launch(void* const* d_in, const int* in_sizes, int n_in,
                              void* d_out, int out_size, void* d_ws, size_t ws_size,
                              hipStream_t stream) {
    const float* x  = (const float*)d_in[0];
    const float* Wq = (const float*)d_in[1];
    const float* Wk = (const float*)d_in[2];
    const float* Wv = (const float*)d_in[3];
    const float* Wo = (const float*)d_in[4];
    const float* Wg = (const float*)d_in[5];
    const float* bg = (const float*)d_in[6];
    const float* Wl = (const float*)d_in[7];
    const float* bl = (const float*)d_in[8];
    float* out = (float*)d_out;

    const size_t NX = (size_t)Bsz * Tsz * Dsz;   // 4M elems
    const size_t NW = (size_t)Dsz * Dsz;         // 1M elems

    __bf16* ws  = (__bf16*)d_ws;
    __bf16* xb  = ws;
    __bf16* wqb = xb + NX;
    __bf16* wkb = wqb + NW;
    __bf16* wvb = wkb + NW;
    __bf16* wob = wvb + NW;
    __bf16* qb  = wob + NW;
    __bf16* kb  = qb + NX;
    __bf16* vb  = kb + NX;
    __bf16* yb  = vb + NX;

    // converts
    cvt_f32_bf16<<<(int)((NX + 255) / 256), 256, 0, stream>>>(x,  xb,  (int)NX);
    cvt_f32_bf16<<<(int)((NW + 255) / 256), 256, 0, stream>>>(Wq, wqb, (int)NW);
    cvt_f32_bf16<<<(int)((NW + 255) / 256), 256, 0, stream>>>(Wk, wkb, (int)NW);
    cvt_f32_bf16<<<(int)((NW + 255) / 256), 256, 0, stream>>>(Wv, wvb, (int)NW);
    cvt_f32_bf16<<<(int)((NW + 255) / 256), 256, 0, stream>>>(Wo, wob, (int)NW);

    const int M = Bsz * Tsz, N = Dsz, K = Dsz;
    dim3 blk(256), grd(N / 128, M / 128);   // (8, 32)
    gemm_bt_bf16<false><<<grd, blk, 0, stream>>>(xb, wqb, nullptr, qb, M, N, K);
    gemm_bt_bf16<false><<<grd, blk, 0, stream>>>(xb, wkb, nullptr, kb, M, N, K);
    gemm_bt_bf16<false><<<grd, blk, 0, stream>>>(xb, wvb, nullptr, vb, M, N, K);

    recurrence_kernel<<<Bsz * Hsz, 32, 0, stream>>>(qb, kb, vb, Wg, bg, Wl, bl, yb);

    gemm_bt_bf16<true><<<grd, blk, 0, stream>>>(yb, wob, out, nullptr, M, N, K);
}
